// LSTMAutoencoder_30253749633447
// MI455X (gfx1250) — matrix operand order repaired and checked by reading, not compiled, not run on hardware
//
#include <hip/hip_runtime.h>

// ---------------------------------------------------------------------------
// LSTM autoencoder, B=512, T=256, D=H=64 for every layer (4H = 256 gates).
//   Per layer: xg = X @ W_ih^T + (b_ih + b_hh)    (big parallel WMMA GEMM)
//              sequential scan over T with h @ W_hh^T done in-registers/LDS.
// bf16 WMMA (f32 accum) for all matmuls; fp32 state & gate math.
// Scan uses async global->LDS double buffering for xg (ASYNCcnt-tracked)
// and gfx1250 hardware V_TANH_F32 for the gate nonlinearities.
// ---------------------------------------------------------------------------

typedef __attribute__((ext_vector_type(16))) __bf16 v16bf;
typedef __attribute__((ext_vector_type(8)))  __bf16 v8bf;
typedef __attribute__((ext_vector_type(8)))  float  v8f;

// 16-byte int vector type required by the async-LDS builtins (b128 flavor).
typedef int v4i __attribute__((vector_size(16)));
typedef __attribute__((address_space(1))) v4i gv4i;  // global-side operand
typedef __attribute__((address_space(3))) v4i lv4i;  // LDS-side operand

#if defined(__has_builtin)
#  if __has_builtin(__builtin_amdgcn_global_load_async_to_lds_b128) && \
      __has_builtin(__builtin_amdgcn_s_wait_asynccnt)
#    define USE_ASYNC_XG 1
#  endif
#endif
#ifndef USE_ASYNC_XG
#  define USE_ASYNC_XG 0
#endif

union FragU { v16bf v; v8bf h[2]; };

static __device__ inline v8f zero8() {
  v8f z = {0.f, 0.f, 0.f, 0.f, 0.f, 0.f, 0.f, 0.f};
  return z;
}

// Hardware tanh (gfx1250 V_TANH_F32) with portable fallback.
static __device__ inline float hw_tanh(float x) {
#if defined(__has_builtin) && __has_builtin(__builtin_amdgcn_tanhf)
  return __builtin_amdgcn_tanhf(x);
#elif defined(__has_builtin) && __has_builtin(__builtin_amdgcn_tanh_f32)
  return __builtin_amdgcn_tanh_f32(x);
#else
  float e = __expf(2.f * x);
  return (e - 1.f) / (e + 1.f);
#endif
}

// sigmoid(x) = 0.5*tanh(x/2) + 0.5  (one trans op + fma on gfx1250)
static __device__ inline float hw_sigmoid(float x) {
  return __builtin_fmaf(0.5f, hw_tanh(0.5f * x), 0.5f);
}

// A fragment: 16x32 bf16 tile, row-major source with ld = 64 elements.
// Lanes 0-15 hold row (lane&15), K = {0..7} in v0..3, {16..23} in v4..7;
// lanes 16-31 shift K by +8. Two 16-byte contiguous loads per lane.
static __device__ inline v16bf load_fragA(const __bf16* base, int lane) {
  int r  = lane & 15;
  int kh = lane >> 4;
  const __bf16* p = base + r * 64 + kh * 8;
  FragU u;
  u.h[0] = *(const v8bf*)(p);
  u.h[1] = *(const v8bf*)(p + 16);
  return u.v;
}

// B fragment: B[k][n] = W[n][k] with W row-major [256][64] bf16.
// Lanes 0-15 hold col (lane&15), K = {0..7} in v0..3, {16..23} in v4..7;
// lanes 16-31 shift K by +8. Two 16-byte contiguous loads per lane.
static __device__ inline v16bf load_fragB_WT(const __bf16* W, int n0, int k0, int lane) {
  int n  = n0 + (lane & 15);
  int kb = k0 + ((lane >> 4) << 3);
  const __bf16* p = W + n * 64 + kb;
  FragU u;
  u.h[0] = *(const v8bf*)(p);
  u.h[1] = *(const v8bf*)(p + 16);
  return u.v;
}

static __device__ inline v8f wmma_bf16(v16bf a, v16bf b, v8f c) {
  return __builtin_amdgcn_wmma_f32_16x16x32_bf16(false, a, false, b,
                                                 (short)0, c, false, false);
}

// ---------------------------------------------------------------------------
// fp32 -> bf16 conversion (inputs & weights)
// ---------------------------------------------------------------------------
__global__ void __launch_bounds__(256)
f32_to_bf16_kernel(const float* __restrict__ in, __bf16* __restrict__ out, int n) {
  int i = blockIdx.x * blockDim.x + threadIdx.x;
  if (i < n) out[i] = (__bf16)in[i];
}

// ---------------------------------------------------------------------------
// xg[m, 0:256] = A[m, 0:64] @ Wih^T + (bih + bhh),  m = b*T + t, M = 131072.
// One wave per 16-row panel; 2 K-chunks x 16 N-tiles = 32 WMMAs/wave.
// ---------------------------------------------------------------------------
__global__ void __launch_bounds__(256)
gemm_xg_kernel(const __bf16* __restrict__ A, const __bf16* __restrict__ Wih,
               const float* __restrict__ bih, const float* __restrict__ bhh,
               float* __restrict__ xg) {
  const int lane = threadIdx.x & 31;
  const int wid  = blockIdx.x * (blockDim.x >> 5) + (threadIdx.x >> 5);
  const int m0   = wid << 4;

  v8f acc[16];
#pragma unroll
  for (int nt = 0; nt < 16; ++nt) acc[nt] = zero8();

  v16bf a0 = load_fragA(A + (size_t)m0 * 64, lane);       // K = 0..31
  v16bf a1 = load_fragA(A + (size_t)m0 * 64 + 32, lane);  // K = 32..63
  v16bf b0 = load_fragB_WT(Wih, 0, 0,  lane);
  v16bf b1 = load_fragB_WT(Wih, 0, 32, lane);

#pragma unroll
  for (int nt = 0; nt < 16; ++nt) {
    v16bf c0 = b0, c1 = b1;
    if (nt < 15) {                      // prefetch next N-tile's fragments
      b0 = load_fragB_WT(Wih, (nt + 1) * 16, 0,  lane);
      b1 = load_fragB_WT(Wih, (nt + 1) * 16, 32, lane);
    }
    acc[nt] = wmma_bf16(a0, c0, acc[nt]);
    acc[nt] = wmma_bf16(a1, c1, acc[nt]);
  }

  const int n = lane & 15, half = lane >> 4;
#pragma unroll
  for (int nt = 0; nt < 16; ++nt) {
    int   col  = nt * 16 + n;
    float bias = bih[col] + bhh[col];
#pragma unroll
    for (int v = 0; v < 8; ++v) {
      int m = v + 8 * half;
      xg[(size_t)(m0 + m) * 256 + col] = acc[nt][v] + bias;
    }
  }
}

#if USE_ASYNC_XG
// Stage xg[b0:b0+16, t, 0:256] (16 KB) into LDS with async b128 copies.
// 1024 16-byte chunks, 512 threads x 2 chunks. Tracked by ASYNCcnt.
static __device__ inline void issue_xg_async(const float* __restrict__ xg,
                                             int b0, int t, float* buf, int tid) {
#pragma unroll
  for (int s = 0; s < 2; ++s) {
    int c = tid + s * 512;
    int m = c >> 6;        // batch row within tile
    int q = c & 63;        // 16-byte chunk within row
    const float* g = xg + ((size_t)(b0 + m) * 256 + t) * 256 + q * 4;
    float*       l = buf + m * 256 + q * 4;
    __builtin_amdgcn_global_load_async_to_lds_b128(
        (gv4i*)g, (lv4i*)l, 0, 0);
  }
}
#endif

// ---------------------------------------------------------------------------
// Recurrent scan. grid = 32 blocks (16 batch rows each), block = 512 = 16 waves.
// Wave w owns gate columns [16w, 16w+16): its W_hh B-fragments live in VGPRs
// for the whole T loop. h tile (16x64 bf16) lives in LDS as the WMMA A operand;
// c lives in registers (2 elements / thread). xg is double-buffered in LDS via
// async copies so global latency is off the recurrence critical path.
// ---------------------------------------------------------------------------
__global__ void __launch_bounds__(512)
lstm_scan_kernel(const float* __restrict__ xg, const __bf16* __restrict__ Whh,
                 __bf16* __restrict__ hout_bf, float* __restrict__ hout_f32) {
  __shared__ __align__(16) __bf16 hTile[16 * 64];   // h_t (bf16), A operand
  __shared__ float gTile[16 * 256];                 // gate pre-activations
#if USE_ASYNC_XG
  __shared__ __align__(16) float xgBuf[2][16 * 256];
#endif

  const int tid  = threadIdx.x;
  const int lane = tid & 31;
  const int w    = tid >> 5;            // wave 0..15 -> gate N-tile
  const int b0   = blockIdx.x << 4;     // first batch row of this block

  // Preload this wave's W_hh fragments (stay in VGPRs for all 256 steps).
  v16bf bw0 = load_fragB_WT(Whh, w * 16, 0,  lane);
  v16bf bw1 = load_fragB_WT(Whh, w * 16, 32, lane);

  for (int i = tid; i < 16 * 64; i += 512) hTile[i] = (__bf16)0.f;

  // Gate-update element mapping: 512 threads x 2 = 16 rows x 64 hidden.
  const int e0 = tid, e1 = tid + 512;
  const int r0 = e0 >> 6, j0 = e0 & 63;
  const int r1 = e1 >> 6, j1 = e1 & 63;
  float c0 = 0.f, c1 = 0.f;

  const int n = lane & 15, half = lane >> 4;
  const int col = w * 16 + n;

#if USE_ASYNC_XG
  issue_xg_async(xg, b0, 0, xgBuf[0], tid);       // stage t = 0
  __builtin_amdgcn_s_wait_asynccnt(0);
#endif
  __syncthreads();

  for (int t = 0; t < 256; ++t) {
#if USE_ASYNC_XG
    const int cur = t & 1;
    if (t + 1 < 256)                               // overlap t+1 copy with step t
      issue_xg_async(xg, b0, t + 1, xgBuf[cur ^ 1], tid);
#endif

    // ---- recurrent GEMM: g = h @ W_hh^T (16x256 per block) ----
    v8f acc = zero8();
    v16bf a0 = load_fragA((const __bf16*)hTile, lane);       // K = 0..31
    v16bf a1 = load_fragA((const __bf16*)hTile + 32, lane);  // K = 32..63
    acc = wmma_bf16(a0, bw0, acc);
    acc = wmma_bf16(a1, bw1, acc);

#pragma unroll
    for (int v = 0; v < 8; ++v) {
      int m = v + 8 * half;
#if USE_ASYNC_XG
      gTile[m * 256 + col] = acc[v] + xgBuf[cur][m * 256 + col];
#else
      size_t row = (size_t)(b0 + m) * 256 + t;   // (b, t) row of xg
      gTile[m * 256 + col] = acc[v] + xg[row * 256 + col];
      __builtin_prefetch(xg + row * 256 + col + 256, 0, 1);  // next timestep
#endif
    }
    __syncthreads();

    // ---- gates + state update (i, f, g, o order); V_TANH_F32 path ----
    {
      float i = hw_sigmoid(gTile[r0 * 256 + j0]);
      float f = hw_sigmoid(gTile[r0 * 256 + 64 + j0]);
      float g = hw_tanh(gTile[r0 * 256 + 128 + j0]);
      float o = hw_sigmoid(gTile[r0 * 256 + 192 + j0]);
      c0 = f * c0 + i * g;
      float h = o * hw_tanh(c0);
      hTile[r0 * 64 + j0] = (__bf16)h;
      size_t orow = (size_t)(b0 + r0) * 256 + t;
      if (hout_bf)  hout_bf[orow * 64 + j0] = (__bf16)h;
      if (hout_f32) hout_f32[orow * 64 + j0] = h;
    }
    {
      float i = hw_sigmoid(gTile[r1 * 256 + j1]);
      float f = hw_sigmoid(gTile[r1 * 256 + 64 + j1]);
      float g = hw_tanh(gTile[r1 * 256 + 128 + j1]);
      float o = hw_sigmoid(gTile[r1 * 256 + 192 + j1]);
      c1 = f * c1 + i * g;
      float h = o * hw_tanh(c1);
      hTile[r1 * 64 + j1] = (__bf16)h;
      size_t orow = (size_t)(b0 + r1) * 256 + t;
      if (hout_bf)  hout_bf[orow * 64 + j1] = (__bf16)h;
      if (hout_f32) hout_f32[orow * 64 + j1] = h;
    }

#if USE_ASYNC_XG
    __builtin_amdgcn_s_wait_asynccnt(0);   // t+1 tile landed (overlapped above)
#endif
    __syncthreads();  // h_t (and xg tile) visible before next step
  }
}

// ---------------------------------------------------------------------------
// Host launcher
// ---------------------------------------------------------------------------
extern "C" void kernel_launch(void* const* d_in, const int* in_sizes, int n_in,
                              void* d_out, int out_size, void* d_ws, size_t ws_size,
                              hipStream_t stream) {
  (void)in_sizes; (void)n_in; (void)out_size; (void)ws_size;
  const int BT = 512 * 256;  // 131072 (b, t) rows

  char*   ws   = (char*)d_ws;
  float*  xg   = (float*)ws;                                    // 134,217,728 B
  __bf16* bufA = (__bf16*)(ws + (size_t)134217728);             //  16,777,216 B
  __bf16* bufB = (__bf16*)(ws + (size_t)134217728 + 16777216);  //  16,777,216 B
  __bf16* wbf  = (__bf16*)(ws + (size_t)134217728 + 2u * 16777216);  // 8x32 KB

  const float* x = (const float*)d_in[0];

  // Convert activations (layer-0 input) and all weights to bf16.
  {
    int n = BT * 64;
    f32_to_bf16_kernel<<<(n + 255) / 256, 256, 0, stream>>>(x, bufA, n);
  }
  for (int L = 0; L < 4; ++L) {
    const float* Wih = (const float*)d_in[1 + 4 * L + 0];
    const float* Whh = (const float*)d_in[1 + 4 * L + 1];
    int n = 256 * 64;
    f32_to_bf16_kernel<<<(n + 255) / 256, 256, 0, stream>>>(Wih, wbf + (2 * L) * 16384, n);
    f32_to_bf16_kernel<<<(n + 255) / 256, 256, 0, stream>>>(Whh, wbf + (2 * L + 1) * 16384, n);
  }

  __bf16* cur = bufA;
  __bf16* nxt = bufB;
  for (int L = 0; L < 4; ++L) {
    const float* bih = (const float*)d_in[1 + 4 * L + 2];
    const float* bhh = (const float*)d_in[1 + 4 * L + 3];

    // 8192 waves of 16 rows each, 8 waves per 256-thread block.
    gemm_xg_kernel<<<BT / 128, 256, 0, stream>>>(cur, wbf + (2 * L) * 16384,
                                                 bih, bhh, xg);
    if (L < 3) {
      lstm_scan_kernel<<<32, 512, 0, stream>>>(xg, wbf + (2 * L + 1) * 16384,
                                               nxt, nullptr);
    } else {
      lstm_scan_kernel<<<32, 512, 0, stream>>>(xg, wbf + (2 * L + 1) * 16384,
                                               nullptr, (float*)d_out);
    }
    __bf16* tmp = cur; cur = nxt; nxt = tmp;
  }
}
